// TransferFunctionModel_15324443312180
// MI455X (gfx1250) — matrix operand order repaired and checked by reading, not compiled, not run on hardware
//
#include <hip/hip_runtime.h>
#include <hip/hip_bf16.h>
#include <math.h>

// ---------------- model constants ----------------
#define BATCH   8
#define CH      128
#define NS      32768
#define KBANK   512
#define NEVENTS 32
#define NATOMS  512
#define ATOMSZ  512
#define NTRANS  64
#define TRSZ    8192
#define NROOMS  8
#define NEVT    (BATCH * NEVENTS)   /* 256 */

typedef __attribute__((ext_vector_type(16))) __bf16 v16bf;
typedef __attribute__((ext_vector_type(8)))  float  v8f;

// float -> bf16 (RNE) as raw u16, and raw u16 -> __bf16 bitcast
__device__ __forceinline__ unsigned short f2bu(float f) {
  union { float f; unsigned u; } c; c.f = f;
  unsigned r = c.u + 0x7FFFu + ((c.u >> 16) & 1u);
  return (unsigned short)(r >> 16);
}
__device__ __forceinline__ __bf16 u2b(unsigned short s) {
  union { unsigned short u; __bf16 b; } c; c.u = s; return c.b;
}
__device__ __forceinline__ __bf16 fb(float x) { return u2b(f2bu(x)); }
__device__ __forceinline__ v8f v8zero() {
  v8f z = {0.f,0.f,0.f,0.f,0.f,0.f,0.f,0.f}; return z;
}
#define WMMA_BF16(A,Bm,Cm) \
  __builtin_amdgcn_wmma_f32_16x16x32_bf16(false,(A),false,(Bm),(short)0,(Cm),false,false)

// ---------------- utility kernels ----------------
__global__ void zero_kernel(float* p, int n) {
  int i = blockIdx.x * 256 + threadIdx.x;
  if (i < n) p[i] = 0.f;
}

// convert filter bank and dilated-conv weights to bf16 (k-major repack for w_net)
__global__ void cvt_weights(const float* __restrict__ bank,
                            const float* __restrict__ wnet,
                            unsigned short* __restrict__ bankb,
                            unsigned short* __restrict__ wnetb) {
  int i = blockIdx.x * 256 + threadIdx.x;
  if (i < CH * KBANK) bankb[i] = f2bu(bank[i]);          // [C][512] same layout
  if (i < 5 * CH * 384) {
    int l = i / (CH * 384);
    int rem = i % (CH * 384);
    int c = rem / 384;
    int k = rem % 384;            // k = j*128 + ci
    int ci = k & 127, j = k >> 7;
    wnetb[i] = f2bu(wnet[(((size_t)l * CH + c) * CH + ci) * 3 + j]);
  }
}

// ---------------- front filter-bank conv (WMMA bf16) ----------------
// h[b,c,t] = sum_k bank[c,k] * x[b, t+k-256]
__global__ void front_conv(const float* __restrict__ x,
                           const unsigned short* __restrict__ bankb,
                           float* __restrict__ h) {
  const int blkPerB = NS / 64;
  int b  = blockIdx.x / blkPerB;
  int t0 = (blockIdx.x % blkPerB) * 64;
  __shared__ unsigned short sx[64 + KBANK];   // bf16 window x[t0-256 .. t0+319]
  for (int j = threadIdx.x; j < 64 + KBANK; j += blockDim.x) {
    int gi = t0 - 256 + j;
    float v = (gi >= 0 && gi < NS) ? x[(size_t)b * NS + gi] : 0.f;
    sx[j] = f2bu(v);
  }
  __syncthreads();
  int lane = threadIdx.x & 31, wave = threadIdx.x >> 5;
  int g = lane >> 4, ln = lane & 15;
  int lt = wave * 16 + ln;                    // local output column 0..63
  v8f acc[8];
#pragma unroll
  for (int mt = 0; mt < 8; ++mt) acc[mt] = v8zero();
  for (int kc = 0; kc < KBANK; kc += 32) {
    v16bf bfr;
#pragma unroll
    for (int i = 0; i < 8; ++i) {
      int k = kc + (((i & 4) << 2) | (g << 3) | ((i & 3) << 1));
      bfr[2*i]   = u2b(sx[lt + k]);
      bfr[2*i+1] = u2b(sx[lt + k + 1]);
    }
#pragma unroll
    for (int mt = 0; mt < 8; ++mt) {
      v16bf afr;
      const unsigned short* ap = bankb + (size_t)(mt * 16 + ln) * KBANK;
#pragma unroll
      for (int i = 0; i < 8; ++i) {
        int k = kc + (((i & 4) << 2) | (g << 3) | ((i & 3) << 1));
        afr[2*i]   = u2b(ap[k]);
        afr[2*i+1] = u2b(ap[k + 1]);
      }
      acc[mt] = WMMA_BF16(afr, bfr, acc[mt]);
    }
  }
  int tc = t0 + lt;
#pragma unroll
  for (int mt = 0; mt < 8; ++mt)
#pragma unroll
    for (int i = 0; i < 8; ++i) {
      int c = mt * 16 + i + 8 * g;
      h[((size_t)b * CH + c) * NS + tc] = acc[mt][i];
    }
}

// ---------------- residual dilated conv layer (WMMA bf16) ----------------
// hout = hin + leaky( sum_{ci,j} w[c,ci,j] * hin[ci, t+(j-1)d] + bias[c] )
__global__ void dilated_conv(const float* __restrict__ hin,
                             const unsigned short* __restrict__ wnetb,
                             const float* __restrict__ bnet,
                             float* __restrict__ hout,
                             int d, int layer) {
  const int blkPerB = NS / 64;
  int b  = blockIdx.x / blkPerB;
  int t0 = (blockIdx.x % blkPerB) * 64;
  int Wd = 64 + 2 * d;                        // <= 118
  __shared__ unsigned short sh[128 * 120];    // bf16 h tile [128][Wd]
  int tot = CH * Wd;
  for (int j = threadIdx.x; j < tot; j += blockDim.x) {
    int ci = j / Wd, col = j % Wd;
    int t = t0 - d + col;
    float v = (t >= 0 && t < NS) ? hin[((size_t)b * CH + ci) * NS + t] : 0.f;
    sh[ci * Wd + col] = f2bu(v);
  }
  __syncthreads();
  int lane = threadIdx.x & 31, wave = threadIdx.x >> 5;
  int g = lane >> 4, ln = lane & 15;
  int lt = wave * 16 + ln;
  const unsigned short* wl = wnetb + (size_t)layer * CH * 384;
  v8f acc[8];
#pragma unroll
  for (int mt = 0; mt < 8; ++mt) acc[mt] = v8zero();
  for (int kc = 0; kc < 384; kc += 32) {
    int jw = kc >> 7;                         // constant over chunk
    int col = lt + jw * d;
    v16bf bfr;
#pragma unroll
    for (int i = 0; i < 8; ++i) {
      int k  = kc + (((i & 4) << 2) | (g << 3) | ((i & 3) << 1));
      int ci = k & 127;
      bfr[2*i]   = u2b(sh[ci * Wd + col]);
      bfr[2*i+1] = u2b(sh[(ci + 1) * Wd + col]);
    }
#pragma unroll
    for (int mt = 0; mt < 8; ++mt) {
      v16bf afr;
      const unsigned short* ap = wl + (size_t)(mt * 16 + ln) * 384;
#pragma unroll
      for (int i = 0; i < 8; ++i) {
        int k = kc + (((i & 4) << 2) | (g << 3) | ((i & 3) << 1));
        afr[2*i]   = u2b(ap[k]);
        afr[2*i+1] = u2b(ap[k + 1]);
      }
      acc[mt] = WMMA_BF16(afr, bfr, acc[mt]);
    }
  }
  int tc = t0 + lt;
#pragma unroll
  for (int mt = 0; mt < 8; ++mt)
#pragma unroll
    for (int i = 0; i < 8; ++i) {
      int c = mt * 16 + i + 8 * g;
      float cv = acc[mt][i] + bnet[layer * CH + c];
      float av = cv > 0.f ? cv : 0.2f * cv;
      size_t o = ((size_t)b * CH + c) * NS + tc;
      hout[o] = hin[o] + av;
    }
}

// ---------------- per-example std reduction ----------------
__global__ void stats_kernel(const float* __restrict__ h,
                             float* __restrict__ sums, float* __restrict__ sumsq) {
  int b = blockIdx.x >> 10;
  int chunk = blockIdx.x & 1023;
  size_t base = (size_t)b * CH * NS + (size_t)chunk * 4096;
  float s = 0.f, q = 0.f;
  for (int j = threadIdx.x; j < 4096; j += 256) {
    float v = h[base + j]; s += v; q += v * v;
  }
  __shared__ float ss[256], sq[256];
  ss[threadIdx.x] = s; sq[threadIdx.x] = q; __syncthreads();
  for (int st = 128; st > 0; st >>= 1) {
    if (threadIdx.x < st) { ss[threadIdx.x] += ss[threadIdx.x + st];
                            sq[threadIdx.x] += sq[threadIdx.x + st]; }
    __syncthreads();
  }
  if (threadIdx.x == 0) { atomicAdd(&sums[b], ss[0]); atomicAdd(&sumsq[b], sq[0]); }
}

__global__ void scale_kernel(const float* sums, const float* sumsq, float* scale) {
  int b = threadIdx.x;
  if (b < BATCH) {
    float n = (float)((long long)CH * NS);
    float mean = sums[b] / n;
    float var = sumsq[b] / n - mean * mean;
    var = var > 0.f ? var : 0.f;
    scale[b] = 1.f / (sqrtf(var) + 1e-8f);
  }
}

// context[b,c] = max_t h_norm
__global__ void context_max(const float* __restrict__ h, const float* __restrict__ scale,
                            float* __restrict__ context) {
  int bc = blockIdx.x;                    // b*CH + c
  size_t base = (size_t)bc * NS;
  float m = -3.402823466e38f;
  for (int j = threadIdx.x; j < NS; j += 256) m = fmaxf(m, h[base + j]);
  __shared__ float sm[256];
  sm[threadIdx.x] = m; __syncthreads();
  for (int st = 128; st > 0; st >>= 1) {
    if (threadIdx.x < st) sm[threadIdx.x] = fmaxf(sm[threadIdx.x], sm[threadIdx.x + st]);
    __syncthreads();
  }
  if (threadIdx.x == 0) context[bc] = sm[0] * scale[bc >> 7];
}

// attn[b,t] = sigmoid( scale[b] * sum_c h[b,c,t]*aw[c] + ab )
__global__ void attn_kernel(const float* __restrict__ h, const float* __restrict__ scale,
                            const float* __restrict__ aw, const float* __restrict__ ab,
                            float* __restrict__ attn) {
  int b = blockIdx.x >> 7;
  int t = ((blockIdx.x & 127) << 8) + threadIdx.x;
  float dot = 0.f;
  for (int c = 0; c < CH; ++c) dot += h[((size_t)b * CH + c) * NS + t] * aw[c];
  float z = dot * scale[b] + ab[0];
  attn[(size_t)b * NS + t] = 1.f / (1.f + expf(-z));
}

// iterative top-32 selection per batch row (stable, lowest-index tie-break)
__global__ void topk_kernel(const float* __restrict__ attn, float* __restrict__ sc,
                            float* __restrict__ vals, int* __restrict__ idxb) {
  int b = blockIdx.x, tid = threadIdx.x;
  for (int j = tid; j < NS; j += 1024) sc[(size_t)b * NS + j] = attn[(size_t)b * NS + j];
  __syncthreads();
  __shared__ float sv[1024];
  __shared__ int   si[1024];
  for (int e = 0; e < NEVENTS; ++e) {
    float bv = -3.402823466e38f; int bi = 0x7fffffff;
    for (int j = tid; j < NS; j += 1024) {
      float v = sc[(size_t)b * NS + j];
      if (v > bv || (v == bv && j < bi)) { bv = v; bi = j; }
    }
    sv[tid] = bv; si[tid] = bi; __syncthreads();
    for (int st = 512; st > 0; st >>= 1) {
      if (tid < st) {
        float ov = sv[tid + st]; int oi = si[tid + st];
        if (ov > sv[tid] || (ov == sv[tid] && oi < si[tid])) { sv[tid] = ov; si[tid] = oi; }
      }
      __syncthreads();
    }
    if (tid == 0) {
      vals[b * NEVENTS + e] = sv[0];
      idxb[b * NEVENTS + e] = si[0];
      sc[(size_t)b * NS + si[0]] = -3.402823466e38f;
    }
    __syncthreads();
  }
}

__global__ void latents_kernel(const float* __restrict__ h, const float* __restrict__ scale,
                               const float* __restrict__ vals, const int* __restrict__ idxb,
                               float* __restrict__ lat) {
  int e = blockIdx.x;               // 0..255 (b-major, matches reshape(-1))
  int b = e >> 5;
  int c = threadIdx.x;
  int t = idxb[e];
  lat[(size_t)e * CH + c] = h[((size_t)b * CH + c) * NS + t] * scale[b] * vals[e];
}

// ---------------- generic bf16 WMMA GEMM: C = act(A[MxK] @ W[KxN] + bias) ----------------
// act: 0 none, 1 leaky(0.2), 2 square
__global__ void gemm_wmma(const float* __restrict__ A, const float* __restrict__ W,
                          const float* __restrict__ bias, float* __restrict__ Co,
                          int M, int N, int K, int act) {
  int lane = threadIdx.x & 31, wave = threadIdx.x >> 5;
  int g = lane >> 4, ln = lane & 15;
  int n  = blockIdx.x * 64 + wave * 16 + ln;
  int m0 = blockIdx.y * 16;
  v8f acc = v8zero();
  for (int kc = 0; kc < K; kc += 32) {
    v16bf afr, bfr;
    int m = m0 + ln;
#pragma unroll
    for (int i = 0; i < 8; ++i) {
      int ka = kc + (((i & 4) << 2) | (g << 3) | ((i & 3) << 1));
      float a0 = (m < M && ka     < K) ? A[(size_t)m * K + ka]     : 0.f;
      float a1 = (m < M && ka + 1 < K) ? A[(size_t)m * K + ka + 1] : 0.f;
      afr[2*i] = fb(a0); afr[2*i+1] = fb(a1);
      int kb = kc + (((i & 4) << 2) | (g << 3) | ((i & 3) << 1));
      float b0 = (n < N && kb     < K) ? W[(size_t)kb * N + n]       : 0.f;
      float b1 = (n < N && kb + 1 < K) ? W[(size_t)(kb + 1) * N + n] : 0.f;
      bfr[2*i] = fb(b0); bfr[2*i+1] = fb(b1);
    }
    acc = WMMA_BF16(afr, bfr, acc);
  }
  if (n < N) {
    float bv = bias ? bias[n] : 0.f;
#pragma unroll
    for (int i = 0; i < 8; ++i) {
      int m = m0 + i + 8 * g;
      if (m < M) {
        float v = acc[i] + bv;
        if (act == 1)      v = v > 0.f ? v : 0.2f * v;
        else if (act == 2) v = v * v;
        Co[(size_t)m * N + n] = v;
      }
    }
  }
}

__global__ void softmax_rows(float* __restrict__ p, int n) {
  int r = blockIdx.x, tid = threadIdx.x;
  float* row = p + (size_t)r * n;
  __shared__ float red[64];
  float m = -3.402823466e38f;
  for (int j = tid; j < n; j += 64) m = fmaxf(m, row[j]);
  red[tid] = m; __syncthreads();
  for (int st = 32; st > 0; st >>= 1) {
    if (tid < st) red[tid] = fmaxf(red[tid], red[tid + st]);
    __syncthreads();
  }
  m = red[0]; __syncthreads();
  float s = 0.f;
  for (int j = tid; j < n; j += 64) { float e = expf(row[j] - m); row[j] = e; s += e; }
  red[tid] = s; __syncthreads();
  for (int st = 32; st > 0; st >>= 1) {
    if (tid < st) red[tid] += red[tid + st];
    __syncthreads();
  }
  float inv = 1.f / red[0];
  for (int j = tid; j < n; j += 64) row[j] *= inv;
}

// ---------------- event synthesis: ev = (conv512(atoms,transfers)+atoms_pad)*amp ----------------
__global__ void event_synth(const float* __restrict__ at_rows, const float* __restrict__ tr_rows,
                            const float* __restrict__ amp, float* __restrict__ ev) {
  int e  = blockIdx.y;
  int i0 = blockIdx.x * 256;
  __shared__ float sat[ATOMSZ];
  __shared__ float str[768];
  for (int j = threadIdx.x; j < ATOMSZ; j += 256) sat[j] = at_rows[(size_t)e * ATOMSZ + j];
  for (int j = threadIdx.x; j < 767; j += 256) {
    int ti = i0 - 511 + j;
    str[j] = (ti >= 0 && ti < TRSZ) ? tr_rows[(size_t)e * TRSZ + ti] : 0.f;
  }
  __syncthreads();
  int t = threadIdx.x;
  float s = 0.f;
  for (int jj = 0; jj < ATOMSZ; ++jj) s += sat[jj] * str[t + 511 - jj];
  int i = i0 + t;
  float tot = s + (i < ATOMSZ ? sat[i] : 0.f);
  ev[(size_t)e * TRSZ + i] = tot * amp[e];
}

__global__ void scatter_kernel(const float* __restrict__ ev, const int* __restrict__ idxb,
                               float* __restrict__ buf) {
  int e = blockIdx.y;
  int i = blockIdx.x * 256 + threadIdx.x;
  atomicAdd(&buf[idxb[e] + i], ev[(size_t)e * TRSZ + i]);
}

// ---------------- reverb ----------------
__global__ void reverb_prep(const float* __restrict__ context,
                            const float* __restrict__ room_w, const float* __restrict__ room_b,
                            const float* __restrict__ mix_w, const float* __restrict__ mix_b,
                            float* __restrict__ rw, float* __restrict__ mixv) {
  int tid = threadIdx.x;              // 64 threads: b*8 + r
  int b = tid >> 3, r = tid & 7;
  __shared__ float lg[64];
  float s = 0.f;
  for (int c = 0; c < CH; ++c) s += context[b * CH + c] * room_w[c * NROOMS + r];
  lg[tid] = s + room_b[r];
  __syncthreads();
  float m = -3.402823466e38f;
  for (int k = 0; k < NROOMS; ++k) m = fmaxf(m, lg[b * 8 + k]);
  float den = 0.f;
  for (int k = 0; k < NROOMS; ++k) den += expf(lg[b * 8 + k] - m);
  rw[tid] = expf(lg[tid] - m) / den;
  if (r == 0) {
    float d = 0.f;
    for (int c = 0; c < CH; ++c) d += context[b * CH + c] * mix_w[c];
    mixv[b] = 1.f / (1.f + expf(-(d + mix_b[0])));
  }
}

// convR[r,t] = sum_{s<=t} dry[s] * rooms[r,t-s]   (causal, zero-padded tiles)
__global__ void conv_rooms(const float* __restrict__ buf, const float* __restrict__ rooms,
                           float* __restrict__ convR) {
  int t0 = blockIdx.x * 128;
  int r  = blockIdx.y;
  int lt = threadIdx.x;
  __shared__ float sd[128];
  __shared__ float sr[256];
  float acc = 0.f;
  for (int sc = 0; sc <= t0; sc += 128) {
    sd[lt] = buf[sc + lt];
    int base = t0 - sc - 127;
    for (int j = lt; j < 255; j += 128) {
      int gi = base + j;
      sr[j] = (gi >= 0 && gi < NS) ? rooms[(size_t)r * NS + gi] : 0.f;
    }
    __syncthreads();
    for (int k = 0; k < 128; ++k) acc += sd[k] * sr[lt + 127 - k];
    __syncthreads();
  }
  convR[(size_t)r * NS + t0 + lt] = acc;
}

__global__ void final_mix(const float* __restrict__ buf, const float* __restrict__ convR,
                          const float* __restrict__ rw, const float* __restrict__ mixv,
                          float* __restrict__ out) {
  int b = blockIdx.y;
  int t = blockIdx.x * 256 + threadIdx.x;
  float wet = 0.f;
#pragma unroll
  for (int r = 0; r < NROOMS; ++r) wet += rw[b * NROOMS + r] * convR[(size_t)r * NS + t];
  float m = mixv[b];
  float dry = buf[t];
  out[(size_t)b * NS + t] = dry * (1.f - m) + wet * m;
}

// ---------------- host orchestration ----------------
extern "C" void kernel_launch(void* const* d_in, const int* in_sizes, int n_in,
                              void* d_out, int out_size, void* d_ws, size_t ws_size,
                              hipStream_t stream) {
  (void)in_sizes; (void)n_in; (void)out_size; (void)ws_size;
  const float* x         = (const float*)d_in[0];
  const float* bank      = (const float*)d_in[1];
  const float* w_net     = (const float*)d_in[2];
  const float* b_net     = (const float*)d_in[3];
  const float* attn_w    = (const float*)d_in[4];
  const float* attn_b    = (const float*)d_in[5];
  const float* atoms_hw  = (const float*)d_in[6];
  const float* atoms_hb  = (const float*)d_in[7];
  const float* atoms_ow  = (const float*)d_in[8];
  const float* atoms_ob  = (const float*)d_in[9];
  const float* trans_hw  = (const float*)d_in[10];
  const float* trans_hb  = (const float*)d_in[11];
  const float* trans_ow  = (const float*)d_in[12];
  const float* trans_ob  = (const float*)d_in[13];
  const float* amp_hw    = (const float*)d_in[14];
  const float* amp_hb    = (const float*)d_in[15];
  const float* amp_ow    = (const float*)d_in[16];
  const float* amp_ob    = (const float*)d_in[17];
  const float* atoms_dict    = (const float*)d_in[18];
  const float* transfer_dict = (const float*)d_in[19];
  const float* rooms     = (const float*)d_in[20];
  const float* room_w    = (const float*)d_in[21];
  const float* room_b    = (const float*)d_in[22];
  const float* mix_w     = (const float*)d_in[23];
  const float* mix_b     = (const float*)d_in[24];
  float* out = (float*)d_out;

  // workspace carve-up (256B aligned)
  char* wsb = (char*)d_ws;
  size_t off = 0;
  auto take = [&](size_t bytes) -> char* {
    char* p = wsb + off;
    off = (off + bytes + 255) & ~(size_t)255;
    return p;
  };
  float* hA   = (float*)take((size_t)BATCH * CH * NS * 4);
  float* hB   = (float*)take((size_t)BATCH * CH * NS * 4);
  unsigned short* bankb = (unsigned short*)take((size_t)CH * KBANK * 2);
  unsigned short* wnetb = (unsigned short*)take((size_t)5 * CH * 384 * 2);
  float* sums   = (float*)take(2 * BATCH * 4);   // sums[0..7], sumsq at +BATCH
  float* sumsq  = sums + BATCH;
  float* scale  = (float*)take(BATCH * 4);
  float* attn   = (float*)take((size_t)BATCH * NS * 4);
  float* attnsc = (float*)take((size_t)BATCH * NS * 4);
  float* context= (float*)take(BATCH * CH * 4);
  float* vals   = (float*)take(NEVT * 4);
  int*   idxb   = (int*)  take(NEVT * 4);
  float* lat    = (float*)take((size_t)NEVT * CH * 4);
  float* tmp0   = (float*)take((size_t)NEVT * CH * 4);
  float* tmp1   = (float*)take((size_t)NEVT * CH * 4);
  float* at_log = (float*)take((size_t)NEVT * NATOMS * 4);
  float* tr_log = (float*)take((size_t)NEVT * NTRANS * 4);
  float* ampv   = (float*)take(NEVT * 4);
  float* at_rows= (float*)take((size_t)NEVT * ATOMSZ * 4);
  float* tr_rows= (float*)take((size_t)NEVT * TRSZ * 4);
  float* ev     = (float*)take((size_t)NEVT * TRSZ * 4);
  float* buf    = (float*)take((size_t)2 * NS * 4);
  float* rw     = (float*)take(BATCH * NROOMS * 4);
  float* mixv   = (float*)take(BATCH * 4);
  float* convR  = (float*)take((size_t)NROOMS * NS * 4);

  // init accumulators
  zero_kernel<<<(2 * NS + 255) / 256, 256, 0, stream>>>(buf, 2 * NS);
  zero_kernel<<<1, 256, 0, stream>>>(sums, 2 * BATCH);

  // weight conversion (bank + dilated weights -> bf16)
  cvt_weights<<<(5 * CH * 384 + 255) / 256, 256, 0, stream>>>(bank, w_net, bankb, wnetb);

  // front-end conv, then 5 residual dilated layers (ping-pong hA<->hB)
  front_conv<<<BATCH * (NS / 64), 128, 0, stream>>>(x, bankb, hA);
  const int dil[5] = {1, 3, 9, 27, 1};
  float* cur = hA; float* nxt = hB;
  for (int l = 0; l < 5; ++l) {
    dilated_conv<<<BATCH * (NS / 64), 128, 0, stream>>>(cur, wnetb, b_net, nxt, dil[l], l);
    float* t_ = cur; cur = nxt; nxt = t_;
  }
  // cur == final h (in hB)

  // normalization stats, context, attention, top-k, latents
  stats_kernel<<<BATCH * 1024, 256, 0, stream>>>(cur, sums, sumsq);
  scale_kernel<<<1, 64, 0, stream>>>(sums, sumsq, scale);
  context_max<<<BATCH * CH, 256, 0, stream>>>(cur, scale, context);
  attn_kernel<<<BATCH * (NS / 256), 256, 0, stream>>>(cur, scale, attn_w, attn_b, attn);
  topk_kernel<<<BATCH, 1024, 0, stream>>>(attn, attnsc, vals, idxb);
  latents_kernel<<<NEVT, CH, 0, stream>>>(cur, scale, vals, idxb, lat);

  // MLP heads (bf16 WMMA GEMMs)
  auto gemm = [&](const float* A, const float* W, const float* bias, float* Co,
                  int M, int N, int K, int act) {
    dim3 g((N + 63) / 64, (M + 15) / 16);
    gemm_wmma<<<g, 128, 0, stream>>>(A, W, bias, Co, M, N, K, act);
  };
  auto mlp = [&](const float* hw, const float* hb, const float* ow, const float* ob,
                 int Nout, float* outp, int actOut) {
    gemm(lat,  hw,              hb,          tmp0, NEVT, CH,   CH, 1);
    gemm(tmp0, hw + CH * CH,    hb + CH,     tmp1, NEVT, CH,   CH, 1);
    gemm(tmp1, hw + 2 * CH * CH,hb + 2 * CH, tmp0, NEVT, CH,   CH, 1);
    gemm(tmp0, ow,              ob,          outp, NEVT, Nout, CH, actOut);
  };
  mlp(atoms_hw, atoms_hb, atoms_ow, atoms_ob, NATOMS, at_log, 0);
  softmax_rows<<<NEVT, 64, 0, stream>>>(at_log, NATOMS);
  gemm(at_log, atoms_dict, nullptr, at_rows, NEVT, ATOMSZ, NATOMS, 0);

  mlp(trans_hw, trans_hb, trans_ow, trans_ob, NTRANS, tr_log, 0);
  softmax_rows<<<NEVT, 64, 0, stream>>>(tr_log, NTRANS);
  gemm(tr_log, transfer_dict, nullptr, tr_rows, NEVT, TRSZ, NTRANS, 0);

  mlp(amp_hw, amp_hb, amp_ow, amp_ob, 1, ampv, 2);   // (.+b)^2

  // event synthesis + overlap-add scatter
  event_synth<<<dim3(TRSZ / 256, NEVT), 256, 0, stream>>>(at_rows, tr_rows, ampv, ev);
  scatter_kernel<<<dim3(TRSZ / 256, NEVT), 256, 0, stream>>>(ev, idxb, buf);

  // reverb: room weights/mix, 8 causal room convolutions, final wet/dry mix
  reverb_prep<<<1, 64, 0, stream>>>(context, room_w, room_b, mix_w, mix_b, rw, mixv);
  conv_rooms<<<dim3(NS / 128, NROOMS), 128, 0, stream>>>(buf, rooms, convR);
  final_mix<<<dim3(NS / 256, BATCH), 256, 0, stream>>>(buf, convR, rw, mixv, out);
}
